// GraphConv_22643067585027
// MI455X (gfx1250) — matrix operand order repaired and checked by reading, not compiled, not run on hardware
//
#include <hip/hip_runtime.h>
#include <stdint.h>

// Problem dims (fixed by reference)
#define NN 64
#define CC 128
#define TT 300
#define VV 25
#define KK 3
#define OO (CC * KK)          // 384 conv output channels
#define TV (TT * VV)          // 7500 spatial positions per (n,c)
#define TTILE 4               // t-values per workgroup
#define COLS (TTILE * VV)     // 100 GEMM columns per workgroup
#define COLS_PAD 112          // padded to 7 x 16
#define NTBLK (TT / TTILE)    // 75
#define NBLK (NN * NTBLK)     // 4800 workgroups in main kernel
#define XT_PITCH 136          // 128 + 8 (bank-conflict padding), bf16 elems
#define CS_PITCH 112
#define BN_EPS 1e-5f

typedef __attribute__((ext_vector_type(16))) __bf16 v16bf;
typedef __attribute__((ext_vector_type(8)))  float  v8f;

// Workspace layout (bytes)
#define OFF_WB    0          // 49152 x uint16 (bf16 swizzled W)      -> 98304 B
#define OFF_BEFF  98304      // 3200 f32 (bias pushed through A)      -> +12800
#define OFF_PSUM  111104     // 4800*128 f32 per-block channel sums   -> +2457600
#define OFF_PSQ   2568704    // 4800*128 f32 per-block channel sumsq  -> +2457600
#define OFF_SCALE 5026304    // 128 f32
#define OFF_SHIFT 5026816    // 128 f32

__device__ __forceinline__ uint16_t f2bf(float f) {
  union { float f; uint32_t u; } c; c.f = f;
  uint32_t u = c.u;
  u += 0x7FFFu + ((u >> 16) & 1u);   // round-to-nearest-even
  return (uint16_t)(u >> 16);
}

// ---------------------------------------------------------------------------
// Prep 1: convert W (384x128 f32) into bf16, pre-swizzled into the exact
// per-lane A-fragment layout of V_WMMA_F32_16X16X32_BF16 (ISA 7.12.2):
// lane L (M = L&15, half = L>>4), element e -> K = (e>=8?16:0) + half*8 + (e&7)
// Stored as [mTile(24)][kTile(4)][lane(32)][e(16)] so an A-fragment is one
// contiguous, 32B-aligned v16bf load per lane.
// ---------------------------------------------------------------------------
__global__ void prep_w(const float* __restrict__ W, uint16_t* __restrict__ wb) {
  int idx = blockIdx.x * 256 + threadIdx.x;       // 0 .. 49151
  if (idx >= OO * CC) return;
  int e  = idx & 15;
  int L  = (idx >> 4) & 31;
  int kt = (idx >> 9) & 3;
  int mt = idx >> 11;                             // 0..23
  int half = L >> 4, n16 = L & 15;
  int o    = mt * 16 + n16;
  int kidx = ((e >= 8) ? 16 : 0) + half * 8 + (e & 7);
  int cp   = kt * 32 + kidx;
  wb[idx] = f2bf(W[o * CC + cp]);
}

// ---------------------------------------------------------------------------
// Prep 2: bEff[c][w] = sum_k b[k*C+c] * sum_v A[k][v][w]  (bias folded through
// the graph aggregation), and copy A to the tail of d_out (tuple output #2).
// ---------------------------------------------------------------------------
__global__ void prep_misc(const float* __restrict__ A, const float* __restrict__ b,
                          float* __restrict__ beff, float* __restrict__ outA) {
  int idx = blockIdx.x * 256 + threadIdx.x;
  if (idx < CC * VV) {
    int c = idx / VV, w = idx % VV;
    float acc = 0.f;
    for (int k = 0; k < KK; ++k) {
      float s = 0.f;
      for (int v = 0; v < VV; ++v) s += A[(k * VV + v) * VV + w];
      acc += b[k * CC + c] * s;
    }
    beff[idx] = acc;
  }
  int j = idx - CC * VV;
  if (j >= 0 && j < KK * VV * VV) outA[j] = A[j];
}

// ---------------------------------------------------------------------------
// Main fused kernel: per workgroup = one n, 4 t-values (100 spatial columns).
//  1. Stage x tile (128 x 100) into LDS transposed as bf16 (B-operand).
//  2. For each 16-channel slab (8 slabs): compute the 3 needed conv row-tiles
//     (o = k*128 + c) x 7 column-tiles with bf16 WMMA, K=128 in 4 steps,
//     f32 accumulate, deposit into LDS.
//  3. Aggregate over (k,v) with A (fp32), add bias-effect + residual, ReLU,
//     write z to d_out, and produce deterministic per-block BN partials.
// LDS: 30464 (xT) + 21504 (convS) + 7500 (A) + 2048 (red) = 61516 B < 64 KB.
// ---------------------------------------------------------------------------
__global__ void __launch_bounds__(256) gcn_main(
    const float* __restrict__ x, const float* __restrict__ Aadj,
    const uint16_t* __restrict__ wb, const float* __restrict__ beff,
    float* __restrict__ out, float* __restrict__ psum, float* __restrict__ psq)
{
  __shared__ __align__(16) uint16_t xT[COLS_PAD * XT_PITCH];
  __shared__ float convS[KK * 16 * CS_PITCH];
  __shared__ float As[KK * VV * VV];
  __shared__ float sRed[2][16][16];

  const int tid = threadIdx.x;
  const int bid = blockIdx.x;
  const int n   = bid / NTBLK;
  const int t0  = (bid % NTBLK) * TTILE;

  for (int i = tid; i < KK * VV * VV; i += 256) As[i] = Aadj[i];

  // Stage x tile: bf16, transposed (column-major over c'), padded columns = 0
  const float* xn = x + (size_t)n * CC * TV + (size_t)t0 * VV;
  for (int i = tid; i < CC * COLS_PAD; i += 256) {
    int cp = i / COLS_PAD, col = i % COLS_PAD;
    float v = (col < COLS) ? xn[(size_t)cp * TV + col] : 0.0f;
    xT[col * XT_PITCH + cp] = f2bf(v);
  }
  __syncthreads();

  const int wave = tid >> 5, lane = tid & 31;
  const int half = lane >> 4, n16 = lane & 15;

  for (int ct = 0; ct < 8; ++ct) {          // 16-channel slab
    // ---- conv GEMM: 3 m-tiles x 7 col-tiles, distributed over 8 waves ----
    for (int tile = wave; tile < 21; tile += 8) {
      int km = tile / 7, c7 = tile % 7;
      int mTile = km * 8 + ct;              // o-rows k*128 + [ct*16, ct*16+16)
      v8f acc = {0.f, 0.f, 0.f, 0.f, 0.f, 0.f, 0.f, 0.f};
#pragma unroll
      for (int kt = 0; kt < 4; ++kt) {      // K = 128 = 4 x 32
        v16bf a = *(const v16bf*)(wb + (size_t)((mTile * 4 + kt) * 32 + lane) * 16);
        const uint16_t* bp = &xT[(c7 * 16 + n16) * XT_PITCH + kt * 32 + half * 8];
        union { uint4 q[2]; v16bf v; } ub;
        ub.q[0] = *(const uint4*)bp;        // ds_read_b128 x2, 16B aligned
        ub.q[1] = *(const uint4*)(bp + 16);
        acc = __builtin_amdgcn_wmma_f32_16x16x32_bf16(
            false, a, false, ub.v, (short)0, acc, false, false);
      }
#pragma unroll
      for (int r = 0; r < 8; ++r)           // D layout: M = r + 8*half, N = n16
        convS[(km * 16 + r + half * 8) * CS_PITCH + c7 * 16 + n16] = acc[r];
    }
    __syncthreads();

    // ---- aggregation + residual + relu + deterministic BN partials ----
    const int ci = tid >> 4, l16 = tid & 15;
    const int c  = ct * 16 + ci;
    float s1 = 0.f, s2 = 0.f;
    float*       orow = out + ((size_t)(n * CC + c) * TT + t0) * VV;
    const float* xrow = x   + ((size_t)(n * CC + c) * TT + t0) * VV;
    for (int col = l16; col < COLS; col += 16) {
      int t = col / VV, w = col % VV;
      float acc = beff[c * VV + w];
#pragma unroll
      for (int k = 0; k < KK; ++k) {
        const float* crow = &convS[(k * 16 + ci) * CS_PITCH + t * VV];
        const float* arow = &As[k * VV * VV + w];
#pragma unroll
        for (int v = 0; v < VV; ++v) acc += crow[v] * arow[v * VV];
      }
      float z = acc + xrow[col];
      z = z > 0.f ? z : 0.f;
      orow[col] = z;
      s1 += z; s2 += z * z;
    }
    sRed[0][ci][l16] = s1;
    sRed[1][ci][l16] = s2;
    __syncthreads();
    if (tid < 16) {                         // fixed-order 16-way reduction
      float a = 0.f, bq = 0.f;
#pragma unroll
      for (int j = 0; j < 16; ++j) { a += sRed[0][tid][j]; bq += sRed[1][tid][j]; }
      psum[(size_t)bid * CC + ct * 16 + tid] = a;
      psq [(size_t)bid * CC + ct * 16 + tid] = bq;
    }
    __syncthreads();                        // convS / sRed reuse
  }
}

// ---------------------------------------------------------------------------
// Stats: one block per channel, fixed-order reduction of 4800 block partials.
// Emits scale = gamma * rsqrt(var + eps), shift = beta - mean * scale.
// ---------------------------------------------------------------------------
__global__ void gcn_stats(const float* __restrict__ psum, const float* __restrict__ psq,
                          const float* __restrict__ gamma, const float* __restrict__ beta,
                          float* __restrict__ scale, float* __restrict__ shift) {
  __shared__ float r1[256], r2[256];
  const int c = blockIdx.x, tid = threadIdx.x;
  float a = 0.f, b = 0.f;
  for (int blk = tid; blk < NBLK; blk += 256) {
    a += psum[(size_t)blk * CC + c];
    b += psq [(size_t)blk * CC + c];
  }
  r1[tid] = a; r2[tid] = b;
  __syncthreads();
  for (int s = 128; s > 0; s >>= 1) {
    if (tid < s) { r1[tid] += r1[tid + s]; r2[tid] += r2[tid + s]; }
    __syncthreads();
  }
  if (tid == 0) {
    const float M = (float)NN * (float)TT * (float)VV;   // 480000
    float mean = r1[0] / M;
    float var  = r2[0] / M - mean * mean;                // biased var
    float sc   = gamma[c] * rsqrtf(var + BN_EPS);
    scale[c] = sc;
    shift[c] = beta[c] - mean * sc;
  }
}

// ---------------------------------------------------------------------------
// Normalize d_out in place, float4-vectorized (7500 % 4 == 0 so a float4
// never crosses a channel plane).
// ---------------------------------------------------------------------------
__global__ void gcn_norm(float* __restrict__ out, const float* __restrict__ scale,
                         const float* __restrict__ shift) {
  size_t i4 = (size_t)blockIdx.x * 256 + threadIdx.x;    // float4 index
  int c = (int)((i4 / (TV / 4)) % CC);
  float4* p = (float4*)out + i4;
  float4 z = *p;
  float sc = scale[c], sh = shift[c];
  z.x = z.x * sc + sh;
  z.y = z.y * sc + sh;
  z.z = z.z * sc + sh;
  z.w = z.w * sc + sh;
  *p = z;
}

extern "C" void kernel_launch(void* const* d_in, const int* in_sizes, int n_in,
                              void* d_out, int out_size, void* d_ws, size_t ws_size,
                              hipStream_t stream) {
  (void)in_sizes; (void)n_in; (void)out_size; (void)ws_size;
  const float* x     = (const float*)d_in[0];
  const float* A     = (const float*)d_in[1];
  const float* W     = (const float*)d_in[2];
  const float* b     = (const float*)d_in[3];
  const float* gamma = (const float*)d_in[4];
  const float* beta  = (const float*)d_in[5];
  float* out = (float*)d_out;

  char* ws = (char*)d_ws;
  uint16_t* wb    = (uint16_t*)(ws + OFF_WB);
  float*    beff  = (float*)(ws + OFF_BEFF);
  float*    psum  = (float*)(ws + OFF_PSUM);
  float*    psq   = (float*)(ws + OFF_PSQ);
  float*    scale = (float*)(ws + OFF_SCALE);
  float*    shift = (float*)(ws + OFF_SHIFT);

  prep_w   <<<192, 256, 0, stream>>>(W, wb);                         // 49152 elems
  prep_misc<<<20, 256, 0, stream>>>(A, b, beff, out + (size_t)NN * CC * TV);
  gcn_main <<<NBLK, 256, 0, stream>>>(x, A, wb, beff, out, psum, psq);
  gcn_stats<<<CC, 256, 0, stream>>>(psum, psq, gamma, beta, scale, shift);
  gcn_norm <<<(NN * CC * TV) / 1024, 256, 0, stream>>>(out, scale, shift);
}
